// BaseNet_17059610099772
// MI455X (gfx1250) — matrix operand order repaired and checked by reading, not compiled, not run on hardware
//
#include <hip/hip_runtime.h>
#include <hip/hip_bf16.h>

// ---------------- problem dims ----------------
#define BB   512
#define TT   120
#define XD   65
#define TD   64
#define INP  160      // 129 padded to multiple of 32
#define IN_REAL 129
#define HH   1024
#define GG   3072     // 3*H
#define OUT  33
#define OUTP 48       // padded to multiple of 16

typedef __bf16 bf16_t;
typedef __attribute__((ext_vector_type(8)))  __bf16 v8bf;
typedef __attribute__((ext_vector_type(16))) __bf16 v16bf;
typedef __attribute__((ext_vector_type(8)))  float  v8f;

// ---------------- helpers ----------------
__device__ __forceinline__ v8f splat8(float v) {
    v8f r;
#pragma unroll
    for (int i = 0; i < 8; ++i) r[i] = v;
    return r;
}

__device__ __forceinline__ float sigm(float x) {
    return 1.0f / (1.0f + __expf(-x));
}

__device__ __forceinline__ v8f wmma_bf16(v16bf a, v16bf b, v8f c) {
    // D = A(16x32) * B(32x16) + C, fp32 accumulate
    return __builtin_amdgcn_wmma_f32_16x16x32_bf16(
        /*neg_a=*/false, a, /*neg_b=*/false, b,
        /*c_mod=*/(short)0, c, /*reuse_a=*/false, /*reuse_b=*/false);
}

// A fragment (16x32 bf16, row-major source, `stride` elements per row).
// lanes 0-15: row M=lane,    elems 0..7 = K k+0..7,  elems 8..15 = K k+16..23
// lanes16-31: row M=lane-16, elems 0..7 = K k+8..15, elems 8..15 = K k+24..31
__device__ __forceinline__ v16bf load_a_frag(const bf16_t* __restrict__ base,
                                             int row, int stride, int k, int lane) {
    const bf16_t* p = base + (size_t)row * stride + k + ((lane >> 4) << 3);
    uint4 u0 = *(const uint4*)(p);
    uint4 u1 = *(const uint4*)(p + 16);
    v8bf lo = __builtin_bit_cast(v8bf, u0);
    v8bf hi = __builtin_bit_cast(v8bf, u1);
    return __builtin_shufflevector(lo, hi, 0,1,2,3,4,5,6,7,8,9,10,11,12,13,14,15);
}

// B fragment (32x16 bf16). Column n = lane&15 maps to weight row `wrow`
// (W stored [n][k] row-major). elems 0..7 = K k+(lane>>4)*8+0..7, elems 8..15 = K k+16+(lane>>4)*8+0..7.
__device__ __forceinline__ v16bf load_b_frag(const bf16_t* __restrict__ base,
                                             int wrow, int stride, int k, int lane) {
    const bf16_t* p = base + (size_t)wrow * stride + k + ((lane >> 4) << 3);
    uint4 u0 = *(const uint4*)(p);
    uint4 u1 = *(const uint4*)(p + 16);
    v8bf lo = __builtin_bit_cast(v8bf, u0);
    v8bf hi = __builtin_bit_cast(v8bf, u1);
    return __builtin_shufflevector(lo, hi, 0,1,2,3,4,5,6,7,8,9,10,11,12,13,14,15);
}

__device__ __forceinline__ v16bf cvt16(float4 a, float4 b, float4 c, float4 d) {
    v16bf r;
    r[0]=(__bf16)a.x; r[1]=(__bf16)a.y; r[2]=(__bf16)a.z; r[3]=(__bf16)a.w;
    r[4]=(__bf16)b.x; r[5]=(__bf16)b.y; r[6]=(__bf16)b.z; r[7]=(__bf16)b.w;
    r[8]=(__bf16)c.x; r[9]=(__bf16)c.y; r[10]=(__bf16)c.z; r[11]=(__bf16)c.w;
    r[12]=(__bf16)d.x; r[13]=(__bf16)d.y; r[14]=(__bf16)d.z; r[15]=(__bf16)d.w;
    return r;
}

// On-the-fly fp32 -> bf16 A fragment (for out_proj)
__device__ __forceinline__ v16bf load_a_frag_f32(const float* __restrict__ base,
                                                 int row, int stride, int k, int lane) {
    const float* p = base + (size_t)row * stride + k + ((lane >> 4) << 3);
    float4 f0 = *(const float4*)(p);
    float4 f1 = *(const float4*)(p + 4);
    float4 f2 = *(const float4*)(p + 16);
    float4 f3 = *(const float4*)(p + 20);
    return cvt16(f0, f1, f2, f3);
}

// ---------------- prep kernels ----------------
__global__ void k_cvt_pad(const float* __restrict__ src, bf16_t* __restrict__ dst,
                          int src_rows, int src_cols, int dst_cols, long n) {
    long i = (long)blockIdx.x * blockDim.x + threadIdx.x;
    if (i >= n) return;
    int r = (int)(i / dst_cols);
    int c = (int)(i % dst_cols);
    float v = (r < src_rows && c < src_cols) ? src[(long)r * src_cols + c] : 0.0f;
    dst[i] = (bf16_t)v;
}

// Build bf16 concat input: [b][t][0:64]=task, [64:129]=x, [129:160]=0
__global__ void k_build_xin(const float* __restrict__ x, const float* __restrict__ task,
                            bf16_t* __restrict__ dst, long n) {
    long i = (long)blockIdx.x * blockDim.x + threadIdx.x;
    if (i >= n) return;
    int k  = (int)(i % INP);
    long bt = i / INP;
    int t = (int)(bt % TT);
    int b = (int)(bt / TT);
    float v = 0.0f;
    if (k < TD)           v = task[(long)b * TD + k];
    else if (k < IN_REAL) v = x[((long)b * TT + t) * XD + (k - TD)];
    dst[i] = (bf16_t)v;
}

__global__ void k_init_h(float* __restrict__ h, bf16_t* __restrict__ hb, long n) {
    long i = (long)blockIdx.x * blockDim.x + threadIdx.x;
    if (i >= n) return;
    h[i]  = 0.1f;
    hb[i] = (bf16_t)0.1f;
}

// ---------------- GRU step ----------------
// One wave computes a 32(batch) x 16(gate-unit) tile of all three gates:
// two A fragments share three B fragments -> 6 WMMAs per 5 fragment loads.
// All 8 waves of a block share the same gate tile (same w_hh rows).
// 1024 wave-tiles -> 128 blocks x 256 threads.
__global__ void __launch_bounds__(256)
gru_step(const bf16_t* __restrict__ xin,   // [B][T][INP] bf16
         const bf16_t* __restrict__ whh,   // [3H][H] bf16
         const bf16_t* __restrict__ wih,   // [3H][INP] bf16
         const float*  __restrict__ b_ih,  // [3H]
         const float*  __restrict__ b_hh,  // [3H]
         const bf16_t* __restrict__ hb_src,// [B][H] bf16 (read)
         bf16_t*       __restrict__ hb_dst,// [B][H] bf16 (write)
         float*        __restrict__ h,     // [B][H] fp32 (read+write, element-private)
         float*        __restrict__ rnn_out,// [B][T][H] fp32
         int t) {
    const int lane   = threadIdx.x & 31;
    const int wid    = threadIdx.x >> 5;          // 0..7
    const int n_tile = blockIdx.x & 63;           // 64 tiles over H=1024
    const int m_half = blockIdx.x >> 6;           // 0..1
    const int m_tile = m_half * 8 + wid;          // 0..15 (32-row tiles)
    const int m_base = m_tile * 32;
    const int j      = n_tile * 16 + (lane & 15); // gate unit 0..1023

    const float br_bias = b_ih[j]          + b_hh[j];
    const float bz_bias = b_ih[j + HH]     + b_hh[j + HH];
    v8f acc_r0  = splat8(br_bias),            acc_r1  = splat8(br_bias);
    v8f acc_z0  = splat8(bz_bias),            acc_z1  = splat8(bz_bias);
    v8f acc_xn0 = splat8(b_ih[j + 2 * HH]),   acc_xn1 = splat8(b_ih[j + 2 * HH]);
    v8f acc_hn0 = splat8(b_hh[j + 2 * HH]),   acc_hn1 = splat8(b_hh[j + 2 * HH]);

    const int arow0 = m_base + (lane & 15);
    const int arow1 = arow0 + 16;

    // h * W_hh^T  (K = 1024)
    for (int k = 0; k < HH; k += 32) {
        if (k + 32 < HH) {  // prefetch next K-chunk of the three weight rows
            __builtin_prefetch(whh + (size_t)j * HH            + k + 32, 0, 3);
            __builtin_prefetch(whh + (size_t)(j + HH) * HH     + k + 32, 0, 3);
            __builtin_prefetch(whh + (size_t)(j + 2 * HH) * HH + k + 32, 0, 3);
        }
        v16bf a0 = load_a_frag(hb_src, arow0, HH, k, lane);
        v16bf a1 = load_a_frag(hb_src, arow1, HH, k, lane);
        v16bf br = load_b_frag(whh, j,          HH, k, lane);
        v16bf bz = load_b_frag(whh, j + HH,     HH, k, lane);
        v16bf bn = load_b_frag(whh, j + 2 * HH, HH, k, lane);
        acc_r0  = wmma_bf16(a0, br, acc_r0);
        acc_r1  = wmma_bf16(a1, br, acc_r1);
        acc_z0  = wmma_bf16(a0, bz, acc_z0);
        acc_z1  = wmma_bf16(a1, bz, acc_z1);
        acc_hn0 = wmma_bf16(a0, bn, acc_hn0);
        acc_hn1 = wmma_bf16(a1, bn, acc_hn1);
    }

    // x_t * W_ih^T  (K = 160)
    const bf16_t* xbase = xin + (size_t)t * INP;   // row stride T*INP
#pragma unroll
    for (int k = 0; k < INP; k += 32) {
        v16bf a0 = load_a_frag(xbase, arow0, TT * INP, k, lane);
        v16bf a1 = load_a_frag(xbase, arow1, TT * INP, k, lane);
        v16bf br = load_b_frag(wih, j,          INP, k, lane);
        v16bf bz = load_b_frag(wih, j + HH,     INP, k, lane);
        v16bf bn = load_b_frag(wih, j + 2 * HH, INP, k, lane);
        acc_r0  = wmma_bf16(a0, br, acc_r0);
        acc_r1  = wmma_bf16(a1, br, acc_r1);
        acc_z0  = wmma_bf16(a0, bz, acc_z0);
        acc_z1  = wmma_bf16(a1, bz, acc_z1);
        acc_xn0 = wmma_bf16(a0, bn, acc_xn0);
        acc_xn1 = wmma_bf16(a1, bn, acc_xn1);
    }

    // gates + h update. C/D layout: element e -> row sub_base + e + 8*(lane>>4)
    const int mo = (lane >> 4) << 3;
#pragma unroll
    for (int s = 0; s < 2; ++s) {
        const v8f& ar  = s ? acc_r1  : acc_r0;
        const v8f& az  = s ? acc_z1  : acc_z0;
        const v8f& axn = s ? acc_xn1 : acc_xn0;
        const v8f& ahn = s ? acc_hn1 : acc_hn0;
#pragma unroll
        for (int e = 0; e < 8; ++e) {
            const int m = m_base + s * 16 + e + mo;
            float r = sigm(ar[e]);
            float z = sigm(az[e]);
            float n = fmaxf(axn[e] + r * ahn[e], 0.0f);
            float hp = h[(size_t)m * HH + j];
            float hn = (1.0f - z) * n + z * hp;
            h[(size_t)m * HH + j]                  = hn;
            hb_dst[(size_t)m * HH + j]             = (bf16_t)hn;
            rnn_out[((size_t)m * TT + t) * HH + j] = hn;
        }
    }
}

// ---------------- output projection ----------------
// out = sigmoid(rnn_hid * w_out^T + b_out); M=61440, Npad=48, K=1024
// 32x16 per wave; 8 waves of a block share the same n tile.
// 1920 m-tiles * 3 n-tiles = 5760 waves -> 720 blocks.
__global__ void __launch_bounds__(256)
out_proj(const float* __restrict__ rnn,     // [B*T][H] fp32
         const bf16_t* __restrict__ wout,   // [OUTP][H] bf16 (rows >=33 zero)
         const float* __restrict__ b_out,   // [33]
         float* __restrict__ out) {         // [B*T][33]
    const int lane    = threadIdx.x & 31;
    const int wid     = threadIdx.x >> 5;
    const int n_tile  = blockIdx.x % 3;
    const int m_block = blockIdx.x / 3;          // 0..239
    const int m_tile  = m_block * 8 + wid;       // 0..1919 (32-row tiles)
    const int m_base  = m_tile * 32;
    const int j       = n_tile * 16 + (lane & 15);

    const float bias = (j < OUT) ? b_out[j] : 0.0f;
    v8f acc0 = splat8(bias), acc1 = splat8(bias);
    const int arow0 = m_base + (lane & 15);
    const int arow1 = arow0 + 16;

    for (int k = 0; k < HH; k += 32) {
        v16bf a0 = load_a_frag_f32(rnn, arow0, HH, k, lane);
        v16bf a1 = load_a_frag_f32(rnn, arow1, HH, k, lane);
        v16bf b  = load_b_frag(wout, j, HH, k, lane);
        acc0 = wmma_bf16(a0, b, acc0);
        acc1 = wmma_bf16(a1, b, acc1);
    }

    const int mo = (lane >> 4) << 3;
    if (j < OUT) {
#pragma unroll
        for (int e = 0; e < 8; ++e) {
            const int bt0 = m_base + e + mo;
            out[(size_t)bt0 * OUT + j]        = sigm(acc0[e]);
            out[(size_t)(bt0 + 16) * OUT + j] = sigm(acc1[e]);
        }
    }
}

// ---------------- launch ----------------
extern "C" void kernel_launch(void* const* d_in, const int* in_sizes, int n_in,
                              void* d_out, int out_size, void* d_ws, size_t ws_size,
                              hipStream_t stream) {
    const float* x     = (const float*)d_in[0];  // [512][120][65]
    const float* task  = (const float*)d_in[1];  // [512][64]
    const float* w_ih  = (const float*)d_in[2];  // [3072][129]
    const float* w_hh  = (const float*)d_in[3];  // [3072][1024]
    const float* b_ih  = (const float*)d_in[4];  // [3072]
    const float* b_hh  = (const float*)d_in[5];  // [3072]
    const float* w_out = (const float*)d_in[6];  // [33][1024]
    const float* b_out = (const float*)d_in[7];  // [33]
    (void)in_sizes; (void)n_in; (void)out_size; (void)ws_size;

    // workspace carve-out (≈30 MB total)
    char* ws = (char*)d_ws;
    size_t off = 0;
    auto carve = [&](size_t bytes) -> void* {
        void* p = ws + off;
        off = (off + bytes + 255) & ~(size_t)255;
        return p;
    };
    bf16_t* wihb  = (bf16_t*)carve((size_t)GG * INP * 2);
    bf16_t* whhb  = (bf16_t*)carve((size_t)GG * HH * 2);
    bf16_t* woutb = (bf16_t*)carve((size_t)OUTP * HH * 2);
    bf16_t* xinb  = (bf16_t*)carve((size_t)BB * TT * INP * 2);
    float*  hbuf  = (float*)carve((size_t)BB * HH * 4);
    bf16_t* hb0   = (bf16_t*)carve((size_t)BB * HH * 2);
    bf16_t* hb1   = (bf16_t*)carve((size_t)BB * HH * 2);

    float* out_p = (float*)d_out;
    float* rnn_p = out_p + (size_t)BB * TT * OUT;

    const int TPB = 256;
    {
        long n = (long)GG * INP;
        k_cvt_pad<<<(n + TPB - 1) / TPB, TPB, 0, stream>>>(w_ih, wihb, GG, IN_REAL, INP, n);
    }
    {
        long n = (long)GG * HH;
        k_cvt_pad<<<(n + TPB - 1) / TPB, TPB, 0, stream>>>(w_hh, whhb, GG, HH, HH, n);
    }
    {
        long n = (long)OUTP * HH;
        k_cvt_pad<<<(n + TPB - 1) / TPB, TPB, 0, stream>>>(w_out, woutb, OUT, HH, HH, n);
    }
    {
        long n = (long)BB * TT * INP;
        k_build_xin<<<(n + TPB - 1) / TPB, TPB, 0, stream>>>(x, task, xinb, n);
    }
    {
        long n = (long)BB * HH;
        k_init_h<<<(n + TPB - 1) / TPB, TPB, 0, stream>>>(hbuf, hb0, n);
    }

    // sequential GRU steps: 1024 wave-tiles -> 128 blocks of 8 waves
    for (int t = 0; t < TT; ++t) {
        const bf16_t* src = (t & 1) ? hb1 : hb0;
        bf16_t*       dst = (t & 1) ? hb0 : hb1;
        gru_step<<<128, TPB, 0, stream>>>(xinb, whhb, wihb, b_ih, b_hh,
                                          src, dst, hbuf, rnn_p, t);
    }

    // output projection
    out_proj<<<720, TPB, 0, stream>>>(rnn_p, woutb, b_out, out_p);
}
